// MaskTokenAggregator_36266703847800
// MI455X (gfx1250) — matrix operand order repaired and checked by reading, not compiled, not run on hardware
//
#include <hip/hip_runtime.h>
#include <hip/hip_bf16.h>
#include <math.h>

// ---------------------------------------------------------------------------
// MaskTokenAggregator for MI455X (gfx1250, wave32, WMMA + TDM)
//
// Roofline: pooling streams the 268MB fp32 feature map twice -> ~23us HBM
// floor at 23.3 TB/s; 2.1 GFLOP is trivial for the matrix pipes, so the
// kernels are organized around coalesced streaming, bf16 WMMA with f32
// accumulate, and TDM double-buffered LDS staging for the tokens GEMM.
// ---------------------------------------------------------------------------

typedef __attribute__((ext_vector_type(16))) __bf16 v16bf;
typedef __attribute__((ext_vector_type(8)))  float  v8f;
typedef unsigned int u32x4 __attribute__((ext_vector_type(4)));
typedef int          i32x4 __attribute__((ext_vector_type(4)));
typedef int          i32x8 __attribute__((ext_vector_type(8)));

#define B_DIM 16
#define C_DIM 256
#define H_DIM 128
#define W_DIM 128
#define N_DIM (H_DIM * W_DIM)   // 16384
#define K_TOK 8
#define D_MODEL 256
#define L_SEQ 9                 // 1 cls + 8 tokens
#define NEG_MIN (-3.4028234663852886e38f)

#if defined(__has_builtin)
#if __has_builtin(__builtin_amdgcn_tensor_load_to_lds) && __has_builtin(__builtin_amdgcn_s_wait_tensorcnt)
#define MTA_HAVE_TDM 1
#endif
#endif

__device__ __forceinline__ __bf16 f2bf(float x) {
  unsigned u = __builtin_bit_cast(unsigned, x);
  unsigned short s = (unsigned short)((u + 0x7FFFu + ((u >> 16) & 1u)) >> 16);
  return __builtin_bit_cast(__bf16, s);
}

#ifdef MTA_HAVE_TDM
// Issue a TDM load of a 2D tile fm[row0..row0+15, col0..col0+31] (fp32,
// row stride N_DIM elements) into LDS at lds_off, padding 1 dword after every
// 32 stored dwords so LDS rows stride 33 dwords (bank-conflict-free reads).
// D# layout per CDNA5 ISA ch.8 (group0 128b, group1 256b; groups 2/3 zero).
__device__ __forceinline__ void tdm_load_tile(unsigned lds_off, unsigned long long gaddr) {
  u32x4 g0;
  g0.x = 1u;                                                  // count=1, user D#
  g0.y = lds_off;                                             // LDS byte address
  g0.z = (unsigned)gaddr;                                     // global addr [31:0]
  g0.w = (unsigned)((gaddr >> 32) & 0x01FFFFFFull) | (2u << 30); // [56:32] | type=2
  i32x8 g1;
  g1[0] = (int)((2u << 16) | (1u << 20) | (4u << 22));        // data_size=4B, pad 1dw/32dw
  g1[1] = (int)(((unsigned)N_DIM & 0xFFFFu) << 16);           // tensor_dim0[15:0]
  g1[2] = (int)(((unsigned)N_DIM >> 16) | (((unsigned)C_DIM & 0xFFFFu) << 16)); // dim0 hi | dim1 lo
  g1[3] = (int)(((unsigned)C_DIM >> 16) | (32u << 16));       // dim1 hi | tile_dim0=32
  g1[4] = (int)(16u);                                         // tile_dim1=16, tile_dim2=0
  g1[5] = (int)((unsigned)N_DIM);                             // tensor_dim0_stride lo (elems)
  g1[6] = 0;                                                  // stride0 hi | stride1 lo
  g1[7] = 0;
  i32x4 gz = {0, 0, 0, 0};
#if defined(__clang_major__) && (__clang_major__ >= 23)
  i32x8 gz8 = {0, 0, 0, 0, 0, 0, 0, 0};
  __builtin_amdgcn_tensor_load_to_lds(g0, g1, gz, gz, gz8, 0);
#else
  __builtin_amdgcn_tensor_load_to_lds(g0, g1, gz, gz, 0);
#endif
}
#endif

// ---------------------------------------------------------------------------
// Kernel 0: zero token accumulator + row_has flags (graph-replay safe)
// ---------------------------------------------------------------------------
__global__ void mta_init_kernel(float* __restrict__ tok, int* __restrict__ rowhas) {
  int i = blockIdx.x * 256 + threadIdx.x;
  if (i < B_DIM * K_TOK * D_MODEL) tok[i] = 0.0f;
  if (i < B_DIM) rowhas[i] = 0;
}

// ---------------------------------------------------------------------------
// Kernel 1: nearest-resize mask [B,1,512,512] -> [B,128,128] > 0.5, plus
// per-batch "any true" flag.
// ---------------------------------------------------------------------------
__global__ __launch_bounds__(256)
void mta_mask_kernel(const float* __restrict__ mask, unsigned char* __restrict__ mb,
                     int* __restrict__ rowhas) {
  int idx = blockIdx.x * 256 + threadIdx.x;     // b*N + n, exact multiple
  int b = idx >> 14;
  int n = idx & (N_DIM - 1);
  int y = n >> 7, x = n & 127;
  float mv = mask[((size_t)b * 512 + (size_t)y * 4) * 512 + (size_t)x * 4];
  unsigned char keep = (mv > 0.5f) ? 1 : 0;
  mb[idx] = keep;
  unsigned long long bal = __ballot((int)keep);
  if ((threadIdx.x & 31) == 0 && (unsigned)bal != 0u) atomicOr(&rowhas[b], 1);
}

// ---------------------------------------------------------------------------
// Kernel 2: logits[b,k,n] = (Q[k,:] . fm[b,:,n]) / 16, masked.
// A-fragments (queries) are invariant across n -> hoisted into 64 VGPRs;
// each wave sweeps 4 n-tiles so per-tile vmem is only the streamed B data.
// ---------------------------------------------------------------------------
__global__ __launch_bounds__(256)
void mta_logits_kernel(const float* __restrict__ fm, const float* __restrict__ qt,
                       const unsigned char* __restrict__ mb, const int* __restrict__ rowhas,
                       float* __restrict__ attn) {
  int lane = threadIdx.x & 31;
  int wave = threadIdx.x >> 5;
  int b = blockIdx.y;
  int tile0 = (blockIdx.x * 8 + wave) * 4;      // 4 tiles of 16 n per wave
  int half = lane >> 4;
  int l15 = lane & 15;
  int arow = (l15 < 7) ? l15 : 7;               // D rows 8..15 are don't-care
  const float* fmb = fm + (size_t)b * C_DIM * N_DIM;
  bool use_mask = rowhas[b] != 0;

  v16bf a[8];
#pragma unroll
  for (int ch = 0; ch < 8; ++ch) {
    const float* qrow = qt + arow * C_DIM + ch * 32 + half * 16;
#pragma unroll
    for (int e = 0; e < 8; ++e) {
      a[ch][e]     = f2bf(qrow[e]);
      a[ch][e + 8] = f2bf(qrow[8 + e]);
    }
  }

  for (int t = 0; t < 4; ++t) {
    int n0 = (tile0 + t) * 16;
    v8f acc = {};
#pragma unroll
    for (int ch = 0; ch < 8; ++ch) {
      v16bf bm;
      const float* bp = fmb + (size_t)(ch * 32 + half * 16) * N_DIM + (n0 + l15);
      if (ch + 1 < 8) __builtin_prefetch(bp + (size_t)32 * N_DIM, 0, 0);
#pragma unroll
      for (int e = 0; e < 16; ++e) bm[e] = f2bf(bp[(size_t)e * N_DIM]);
      acc = __builtin_amdgcn_wmma_f32_16x16x32_bf16(false, a[ch], false, bm,
                                                    (short)0, acc, false, false);
    }
    if (half == 0) {
      int n = n0 + l15;
      bool keep = mb[b * N_DIM + n] != 0;
      bool kill = use_mask && !keep;
#pragma unroll
      for (int r = 0; r < K_TOK; ++r) {
        float v = acc[r] * 0.0625f;             // 1/sqrt(256)
        if (kill) v = NEG_MIN;
        attn[(size_t)(b * K_TOK + r) * N_DIM + n] = v;
      }
    }
  }
}

// ---------------------------------------------------------------------------
// Kernel 3: in-place softmax over rows of length N_DIM (128 rows)
// ---------------------------------------------------------------------------
__global__ __launch_bounds__(256)
void mta_softmax_kernel(float* __restrict__ attn) {
  __shared__ float red[256];
  float* p = attn + (size_t)blockIdx.x * N_DIM;
  int tid = threadIdx.x;

  float m = NEG_MIN;
  for (int i = tid; i < N_DIM; i += 256) m = fmaxf(m, p[i]);
  red[tid] = m; __syncthreads();
  for (int s = 128; s > 0; s >>= 1) { if (tid < s) red[tid] = fmaxf(red[tid], red[tid + s]); __syncthreads(); }
  m = red[0]; __syncthreads();

  float sum = 0.0f;
  for (int i = tid; i < N_DIM; i += 256) { float e = __expf(p[i] - m); p[i] = e; sum += e; }
  red[tid] = sum; __syncthreads();
  for (int s = 128; s > 0; s >>= 1) { if (tid < s) red[tid] += red[tid + s]; __syncthreads(); }
  float inv = 1.0f / red[0];
  for (int i = tid; i < N_DIM; i += 256) p[i] *= inv;
}

// ---------------------------------------------------------------------------
// Kernel 4: tokens[b,k,c] += sum_n attn[b,k,n] * fm[b,c,n]  (WMMA, K-split 16)
// TDM path: double-buffered tensor_load_to_lds of the 16x32 fp32 B tile,
// s_wait_tensorcnt pipelining, padded LDS rows (33 dwords) for conflict-free
// per-lane b128 reads. Fallback: direct b128-vectorizable global loads.
// ---------------------------------------------------------------------------
__global__ __launch_bounds__(32)
void mta_tokens_kernel(const float* __restrict__ fm, const float* __restrict__ attn,
                       float* __restrict__ tok) {
  int lane = threadIdx.x & 31;
  int b = blockIdx.z;
  int c0 = blockIdx.y * 16;
  int nbeg = blockIdx.x * (N_DIM / 16);         // 1024-wide K slice
  int half = lane >> 4;
  int l15 = lane & 15;
  int arow = (l15 < 7) ? l15 : 7;
  const float* ar = attn + (size_t)(b * K_TOK + arow) * N_DIM;
  v8f acc = {};

#ifdef MTA_HAVE_TDM
  __shared__ float stage[2][16 * 33];           // padded rows: 33 dwords
  const float* gtile = fm + (size_t)b * C_DIM * N_DIM + (size_t)c0 * N_DIM + nbeg;
  unsigned long long gbase = (unsigned long long)(uintptr_t)gtile;
  const int NCHUNK = (N_DIM / 16) / 32;         // 32 chunks of K=32
  tdm_load_tile((unsigned)(uintptr_t)&stage[0][0], gbase);
  for (int it = 0; it < NCHUNK; ++it) {
    int cur = it & 1;
    if (it + 1 < NCHUNK) {
      tdm_load_tile((unsigned)(uintptr_t)&stage[(it + 1) & 1][0],
                    gbase + (unsigned long long)(it + 1) * 128ull);
      __builtin_amdgcn_s_wait_tensorcnt(1);     // chunk `it` has landed
    } else {
      __builtin_amdgcn_s_wait_tensorcnt(0);
    }
    int n0 = nbeg + it * 32;
    v16bf a, bm;
    const float* ap = ar + n0 + half * 16;
#pragma unroll
    for (int e = 0; e < 8; ++e) {
      a[e]     = f2bf(ap[e]);
      a[e + 8] = f2bf(ap[8 + e]);
    }
    const float* sp = &stage[cur][l15 * 33 + half * 16];
#pragma unroll
    for (int e = 0; e < 16; ++e) bm[e] = f2bf(sp[e]);
    acc = __builtin_amdgcn_wmma_f32_16x16x32_bf16(false, a, false, bm,
                                                  (short)0, acc, false, false);
  }
#else
  const float* brow = fm + (size_t)b * C_DIM * N_DIM + (size_t)(c0 + l15) * N_DIM;
  for (int n0 = nbeg; n0 < nbeg + (N_DIM / 16); n0 += 32) {
    v16bf a, bm;
    const float* ap = ar + n0 + half * 16;
#pragma unroll
    for (int e = 0; e < 8; ++e) {
      a[e]     = f2bf(ap[e]);
      a[e + 8] = f2bf(ap[8 + e]);
    }
    const float* bp = brow + n0 + half * 16;
#pragma unroll
    for (int e = 0; e < 16; ++e) bm[e] = f2bf(bp[e]);
    acc = __builtin_amdgcn_wmma_f32_16x16x32_bf16(false, a, false, bm,
                                                  (short)0, acc, false, false);
  }
#endif

  if (half == 0) {
#pragma unroll
    for (int r = 0; r < K_TOK; ++r)
      atomicAdd(&tok[(b * K_TOK + r) * D_MODEL + c0 + l15], acc[r]);
  }
}

// ---------------------------------------------------------------------------
// Kernel 5: mixer block on x = [cls; tokens] per batch, fp32 in LDS.
// ---------------------------------------------------------------------------
__device__ __forceinline__ void ln_rows(const float* src, float* dst,
                                        const float* w, const float* bb,
                                        float* red, int tid) {
  for (int t = 0; t < L_SEQ; ++t) {
    float v = src[t * D_MODEL + tid];
    red[tid] = v; __syncthreads();
    for (int s = 128; s > 0; s >>= 1) { if (tid < s) red[tid] += red[tid + s]; __syncthreads(); }
    float mu = red[0] * (1.0f / 256.0f); __syncthreads();
    float dv = v - mu;
    red[tid] = dv * dv; __syncthreads();
    for (int s = 128; s > 0; s >>= 1) { if (tid < s) red[tid] += red[tid + s]; __syncthreads(); }
    float var = red[0] * (1.0f / 256.0f); __syncthreads();
    dst[t * D_MODEL + tid] = dv * rsqrtf(var + 1e-5f) * w[tid] + bb[tid];
  }
}

__global__ __launch_bounds__(256)
void mta_mixer_kernel(const float* __restrict__ tok, const float* __restrict__ cls,
                      const float* __restrict__ ln1w, const float* __restrict__ ln1b,
                      const float* __restrict__ ipw,  const float* __restrict__ ipb,
                      const float* __restrict__ opw,  const float* __restrict__ opb,
                      const float* __restrict__ ln2w, const float* __restrict__ ln2b,
                      const float* __restrict__ w1,   const float* __restrict__ b1,
                      const float* __restrict__ w2,   const float* __restrict__ b2,
                      const float* __restrict__ onw,  const float* __restrict__ onb,
                      float* __restrict__ out_cls, float* __restrict__ out_tok) {
  __shared__ float xb[L_SEQ * D_MODEL];
  __shared__ float hb[L_SEQ * D_MODEL];
  __shared__ float big[L_SEQ * 1024];           // qkv (9x768) then mlp u (9x1024)
  __shared__ float sc[4 * L_SEQ * L_SEQ];
  __shared__ float red[256];
  int b = blockIdx.x, tid = threadIdx.x;

  for (int i = tid; i < L_SEQ * D_MODEL; i += 256) {
    int t = i / D_MODEL, d = i % D_MODEL;
    xb[i] = (t == 0) ? cls[d] : tok[(b * K_TOK + (t - 1)) * D_MODEL + d];
  }
  __syncthreads();

  ln_rows(xb, hb, ln1w, ln1b, red, tid);
  __syncthreads();

  float* qkv = big;                             // [t][768]
  for (int i = tid; i < L_SEQ * 768; i += 256) {
    int t = i / 768, j = i % 768;
    const float* w = ipw + (size_t)j * D_MODEL;
    const float* h = hb + t * D_MODEL;
    float s = ipb[j];
    for (int d = 0; d < D_MODEL; ++d) s += h[d] * w[d];
    qkv[i] = s;
  }
  __syncthreads();

  for (int i = tid; i < 4 * L_SEQ * L_SEQ; i += 256) {
    int hh = i / (L_SEQ * L_SEQ);
    int r = i % (L_SEQ * L_SEQ);
    int qi = r / L_SEQ, kj = r % L_SEQ;
    const float* q = qkv + qi * 768 + hh * 64;
    const float* k = qkv + kj * 768 + 256 + hh * 64;
    float s = 0.0f;
    for (int e = 0; e < 64; ++e) s += q[e] * k[e];
    sc[i] = s * 0.125f;                         // 1/sqrt(64)
  }
  __syncthreads();

  if (tid < 4 * L_SEQ) {                        // softmax over 9
    float* row = sc + tid * L_SEQ;
    float m = row[0];
    for (int j = 1; j < L_SEQ; ++j) m = fmaxf(m, row[j]);
    float s = 0.0f;
    for (int j = 0; j < L_SEQ; ++j) { row[j] = __expf(row[j] - m); s += row[j]; }
    float inv = 1.0f / s;
    for (int j = 0; j < L_SEQ; ++j) row[j] *= inv;
  }
  __syncthreads();

  for (int i = tid; i < L_SEQ * D_MODEL; i += 256) {   // o -> hb
    int t = i / D_MODEL, d = i % D_MODEL;
    int hh = d / 64;
    const float* arow = sc + (hh * L_SEQ + t) * L_SEQ;
    float s = 0.0f;
    for (int j = 0; j < L_SEQ; ++j) s += arow[j] * qkv[j * 768 + 512 + d];
    hb[i] = s;
  }
  __syncthreads();

  for (int i = tid; i < L_SEQ * D_MODEL; i += 256) {   // out_proj + residual
    int t = i / D_MODEL, d = i % D_MODEL;
    const float* w = opw + (size_t)d * D_MODEL;
    const float* o = hb + t * D_MODEL;
    float s = opb[d];
    for (int e = 0; e < D_MODEL; ++e) s += o[e] * w[e];
    xb[i] += s;
  }
  __syncthreads();

  ln_rows(xb, hb, ln2w, ln2b, red, tid);
  __syncthreads();

  float* u = big;                               // reuse LDS: [t][1024]
  for (int i = tid; i < L_SEQ * 1024; i += 256) {
    int t = i / 1024, j = i % 1024;
    const float* w = w1 + (size_t)j * D_MODEL;
    const float* h = hb + t * D_MODEL;
    float s = b1[j];
    for (int d = 0; d < D_MODEL; ++d) s += h[d] * w[d];
    u[i] = 0.5f * s * (1.0f + erff(s * 0.70710678118654752f));   // exact gelu
  }
  __syncthreads();

  for (int i = tid; i < L_SEQ * D_MODEL; i += 256) {   // mlp2 + residual
    int t = i / D_MODEL, d = i % D_MODEL;
    const float* w = w2 + (size_t)d * 1024;
    const float* uu = u + t * 1024;
    float s = b2[d];
    for (int e = 0; e < 1024; ++e) s += uu[e] * w[e];
    xb[i] += s;
  }
  __syncthreads();

  ln_rows(xb, hb, onw, onb, red, tid);
  __syncthreads();

  for (int i = tid; i < L_SEQ * D_MODEL; i += 256) {
    int t = i / D_MODEL, d = i % D_MODEL;
    float v = hb[i];
    if (t == 0) out_cls[b * D_MODEL + d] = v;
    else        out_tok[(b * K_TOK + (t - 1)) * D_MODEL + d] = v;
  }
}

// ---------------------------------------------------------------------------
extern "C" void kernel_launch(void* const* d_in, const int* in_sizes, int n_in,
                              void* d_out, int out_size, void* d_ws, size_t ws_size,
                              hipStream_t stream) {
  const float* fm   = (const float*)d_in[0];    // [16,256,128,128]
  const float* mask = (const float*)d_in[1];    // [16,1,512,512]
  const float* qt   = (const float*)d_in[2];    // [8,256]
  const float* cls  = (const float*)d_in[3];    // [1,1,256]
  const float* ln1w = (const float*)d_in[4];
  const float* ln1b = (const float*)d_in[5];
  const float* ipw  = (const float*)d_in[6];
  const float* ipb  = (const float*)d_in[7];
  const float* opw  = (const float*)d_in[8];
  const float* opb  = (const float*)d_in[9];
  const float* ln2w = (const float*)d_in[10];
  const float* ln2b = (const float*)d_in[11];
  const float* w1   = (const float*)d_in[12];
  const float* b1   = (const float*)d_in[13];
  const float* w2   = (const float*)d_in[14];
  const float* b2   = (const float*)d_in[15];
  const float* onw  = (const float*)d_in[16];
  const float* onb  = (const float*)d_in[17];

  // output layout: [cls 16*256 | tokens 16*8*256 | attn 16*8*16384]
  float* out_cls = (float*)d_out;
  float* out_tok = out_cls + B_DIM * D_MODEL;
  float* attn    = out_tok + B_DIM * K_TOK * D_MODEL;

  // workspace: mask bytes | row_has flags | token accumulator
  char* ws = (char*)d_ws;
  unsigned char* mb = (unsigned char*)ws;                         // 262144 B
  int*   rowhas = (int*)(ws + (size_t)B_DIM * N_DIM);             // 64 B
  float* tokacc = (float*)(ws + (size_t)B_DIM * N_DIM + 256);     // 131072 B

  mta_init_kernel<<<128, 256, 0, stream>>>(tokacc, rowhas);
  mta_mask_kernel<<<(B_DIM * N_DIM) / 256, 256, 0, stream>>>(mask, mb, rowhas);
  mta_logits_kernel<<<dim3(N_DIM / (16 * 8 * 4), B_DIM), 256, 0, stream>>>(fm, qt, mb, rowhas, attn);
  mta_softmax_kernel<<<B_DIM * K_TOK, 256, 0, stream>>>(attn);
  mta_tokens_kernel<<<dim3(16, D_MODEL / 16, B_DIM), 32, 0, stream>>>(fm, attn, tokacc);
  mta_mixer_kernel<<<B_DIM, 256, 0, stream>>>(tokacc, cls, ln1w, ln1b, ipw, ipb,
                                              opw, opb, ln2w, ln2b, w1, b1, w2, b2,
                                              onw, onb, out_cls, out_tok);
}
